// LSTMModel_72138270703797
// MI455X (gfx1250) — compile-verified, matrix operand order checked by reading, not run on hardware
//
#include <hip/hip_runtime.h>
#include <hip/hip_bf16.h>
#include <stdint.h>

// ---------------- problem constants ----------------
#define B_   64
#define T_   512
#define I_   256
#define H_   512
#define O_   128
#define G_   (4 * H_)        // 2048 gate rows
#define K_   (I_ + H_)       // 768 merged contraction dim
#define NWG  16              // persistent workgroups
#define NTHR 256             // 8 waves (wave32)
#define HCOLS 32             // h-columns owned per WGP
#define NLOC 128             // gate columns per WGP (4 gates x 32)

// LDS layout (dynamic): padded strides put row stride = 8 mod 64 banks (2-way max)
#define WSTR 784             // padded LDS row stride for weight slice (768+16)
#define HSTR 528             // padded LDS row stride for h stage (512+16)
#define W_LDS_BYTES ((size_t)NLOC * WSTR * 2)   // 200704
#define H_LDS_BYTES ((size_t)B_ * HSTR * 2)     // 67584
#define LDS_TOTAL   (W_LDS_BYTES + H_LDS_BYTES) // 268288 < 320KB WGP LDS

typedef __attribute__((ext_vector_type(16))) __bf16 v16bf;
typedef __attribute__((ext_vector_type(8)))  float  v8f;
typedef int v4i_vs __attribute__((vector_size(16)));   // matches builtin param type

// ---------------- workspace layout (bytes) ----------------
static constexpr size_t OFF_W    = 0;                                   // bf16 [G_][K_]
static constexpr size_t OFF_FCW  = OFF_W    + (size_t)G_ * K_ * 2;      // bf16 [O_][H_]
static constexpr size_t OFF_BSUM = OFF_FCW  + (size_t)O_ * H_ * 2;      // f32  [G_]
static constexpr size_t OFF_XT   = OFF_BSUM + (size_t)G_ * 4;           // bf16 [T_][B_][I_]
static constexpr size_t OFF_HBUF = OFF_XT   + (size_t)T_ * B_ * I_ * 2; // bf16 [2][B_][H_]
static constexpr size_t OFF_BAR  = OFF_HBUF + (size_t)2 * B_ * H_ * 2;  // int

__device__ __forceinline__ unsigned short f2bf(float f) {
    union { float f; uint32_t u; } v; v.f = f;
    uint32_t r = v.u + 0x7fffu + ((v.u >> 16) & 1u);   // round-to-nearest-even
    return (unsigned short)(r >> 16);
}
__device__ __forceinline__ float sigm(float x) { return 1.f / (1.f + __expf(-x)); }

// ---- async global->LDS copy helpers (ASYNCcnt path), guarded by probe ----
#if __has_builtin(__builtin_amdgcn_global_load_async_to_lds_b128) && \
    __has_builtin(__builtin_amdgcn_s_wait_asynccnt)
#define HAS_ASYNC_LDS 1
#else
#define HAS_ASYNC_LDS 0
#endif

__device__ __forceinline__ void copy16_g2l(unsigned short* ldst, const unsigned short* gsrc) {
#if HAS_ASYNC_LDS
    __builtin_amdgcn_global_load_async_to_lds_b128(
        (__attribute__((address_space(1))) v4i_vs*)gsrc,
        (__attribute__((address_space(3))) v4i_vs*)ldst, 0, 0);
#else
    *(uint4*)ldst = *(const uint4*)gsrc;
#endif
}
__device__ __forceinline__ void async_join() {
#if HAS_ASYNC_LDS
    __builtin_amdgcn_s_wait_asynccnt(0);
#endif
}

// ---------------- prep kernels ----------------
__global__ void k_prep_w(const float* __restrict__ Wi, const float* __restrict__ Wh,
                         unsigned short* __restrict__ W) {
    int idx = blockIdx.x * blockDim.x + threadIdx.x;
    if (idx >= G_ * K_) return;
    int g = idx / K_, k = idx % K_;
    float v = (k < I_) ? Wi[g * I_ + k] : Wh[g * H_ + (k - I_)];
    W[idx] = f2bf(v);
}

__global__ void k_prep_misc(const float* __restrict__ bi, const float* __restrict__ bh,
                            const float* __restrict__ fcw_f,
                            unsigned short* __restrict__ fcw, float* __restrict__ bsum,
                            unsigned short* __restrict__ hbuf, int* __restrict__ bar) {
    int idx = blockIdx.x * blockDim.x + threadIdx.x;   // 65536 threads
    if (idx < O_ * H_)     fcw[idx] = f2bf(fcw_f[idx]);
    if (idx < G_)          bsum[idx] = bi[idx] + bh[idx];
    if (idx < 2 * B_ * H_) hbuf[idx] = 0;              // h0 = 0 (re-zeroed every launch)
    if (idx == 0)          *bar = 0;                   // barrier counter reset every launch
}

__global__ void k_prep_x(const float* __restrict__ x, unsigned short* __restrict__ xT) {
    int idx = blockIdx.x * blockDim.x + threadIdx.x;   // B_*T_*I_ threads
    if (idx >= B_ * T_ * I_) return;
    int i = idx % I_;
    int t = (idx / I_) % T_;
    int b = idx / (I_ * T_);
    xT[((size_t)t * B_ + b) * I_ + i] = f2bf(x[idx]);
}

// ---------------- persistent recurrent kernel ----------------
__global__ void __launch_bounds__(NTHR)
k_lstm(const unsigned short* __restrict__ W, const float* __restrict__ bsum,
       const unsigned short* __restrict__ xT, unsigned short* __restrict__ hbuf,
       int* __restrict__ bar, const unsigned short* __restrict__ fcw,
       const float* __restrict__ fcb, float* __restrict__ out) {
    extern __shared__ __align__(16) unsigned char dynsmem[];
    unsigned short* Wlds = (unsigned short*)dynsmem;                  // [128][WSTR] bf16
    unsigned short* hS   = (unsigned short*)(dynsmem + W_LDS_BYTES);  // [64][HSTR] bf16
    float*          gS   = (float*)(dynsmem + W_LDS_BYTES);           // [64][128] f32 (aliases hS)

    const int tid   = threadIdx.x;
    const int lane  = tid & 31;
    const int wave  = tid >> 5;
    const int wg    = blockIdx.x;

    const int mtile = wave & 3;                 // M tile (rows 16*mtile..)
    const int ngrp  = wave >> 2;                // which 64-wide N group
    const int nlane = lane & 15;
    const int khalf = lane >> 4;                // lane half per 16-bit WMMA layout
    const int arow  = mtile * 16 + nlane;       // batch row this lane loads

    // ---- one-time: stage this WGP's weight slice (128 x 768 bf16) into LDS ----
    // local row ln -> global gate row g = (ln/32)*H + wg*32 + (ln%32)
    for (int it = 0; it < (NLOC * (K_ / 8)) / NTHR; ++it) {     // 12288 chunks / 256 thr = 48
        int u   = tid + it * NTHR;
        int ln  = u / (K_ / 8);                 // 0..127
        int j   = u % (K_ / 8);                 // 8-elem chunk in row
        int g   = (ln >> 5) * H_ + wg * HCOLS + (ln & 31);
        copy16_g2l(Wlds + (size_t)ln * WSTR + j * 8, W + (size_t)g * K_ + j * 8);
    }

    // per-lane LDS offsets for the 4 B tiles this wave consumes
    int wofs[4];
#pragma unroll
    for (int nt = 0; nt < 4; ++nt) {
        int ln = ngrp * 64 + nt * 16 + nlane;   // 0..127 local gate col
        wofs[nt] = ln * WSTR + khalf * 16;
    }

    float c[8];
#pragma unroll
    for (int j = 0; j < 8; ++j) c[j] = 0.f;

    async_join();
    __syncthreads();    // weights resident in LDS

    int cur = 0;
    for (int t = 0; t < T_; ++t) {
        // ---- issue async stage of h_cur (64x512 bf16 -> padded LDS rows) ----
        {
            const unsigned short* src = hbuf + (size_t)cur * B_ * H_;
#pragma unroll
            for (int it = 0; it < (B_ * (H_ / 8)) / NTHR; ++it) {   // 4096 chunks / 256 = 16
                int u = tid + it * NTHR;
                int r = u >> 6;                 // row 0..63
                int j = u & 63;                 // 8-elem chunk in row
                copy16_g2l(hS + (size_t)r * HSTR + j * 8, src + (size_t)r * H_ + j * 8);
            }
        }

        if (t + 1 < T_)
            __builtin_prefetch(xT + ((size_t)(t + 1) * B_ + arow) * I_, 0, 1);

        v8f a0 = {}, a1 = {}, a2 = {}, a3 = {};

        // ---- x-phase: K = 0..255, A from global (L2-hot), B from LDS ----
        const unsigned short* xrow = xT + ((size_t)t * B_ + arow) * I_ + khalf * 16;
#pragma unroll 2
        for (int kb = 0; kb < I_ / 32; ++kb) {
            const int K0 = kb * 32;
            v16bf A  = *(const v16bf*)(xrow + K0);
            v16bf b0 = *(const v16bf*)(Wlds + wofs[0] + K0);
            v16bf b1 = *(const v16bf*)(Wlds + wofs[1] + K0);
            v16bf b2 = *(const v16bf*)(Wlds + wofs[2] + K0);
            v16bf b3 = *(const v16bf*)(Wlds + wofs[3] + K0);
            a0 = __builtin_amdgcn_wmma_f32_16x16x32_bf16(false, A, false, b0, (short)0, a0, false, false);
            a1 = __builtin_amdgcn_wmma_f32_16x16x32_bf16(false, A, false, b1, (short)0, a1, false, false);
            a2 = __builtin_amdgcn_wmma_f32_16x16x32_bf16(false, A, false, b2, (short)0, a2, false, false);
            a3 = __builtin_amdgcn_wmma_f32_16x16x32_bf16(false, A, false, b3, (short)0, a3, false, false);
        }

        async_join();       // h stage complete (this wave)
        __syncthreads();    // h stage complete (all waves)

        // ---- h-phase: K = 256..767, A from LDS, B from LDS ----
        const unsigned short* hrow = hS + (size_t)arow * HSTR + khalf * 16;
#pragma unroll 2
        for (int kb = 0; kb < H_ / 32; ++kb) {
            const int K0 = kb * 32;
            const int KW = I_ + K0;             // offset into weight rows
            v16bf A  = *(const v16bf*)(hrow + K0);
            v16bf b0 = *(const v16bf*)(Wlds + wofs[0] + KW);
            v16bf b1 = *(const v16bf*)(Wlds + wofs[1] + KW);
            v16bf b2 = *(const v16bf*)(Wlds + wofs[2] + KW);
            v16bf b3 = *(const v16bf*)(Wlds + wofs[3] + KW);
            a0 = __builtin_amdgcn_wmma_f32_16x16x32_bf16(false, A, false, b0, (short)0, a0, false, false);
            a1 = __builtin_amdgcn_wmma_f32_16x16x32_bf16(false, A, false, b1, (short)0, a1, false, false);
            a2 = __builtin_amdgcn_wmma_f32_16x16x32_bf16(false, A, false, b2, (short)0, a2, false, false);
            a3 = __builtin_amdgcn_wmma_f32_16x16x32_bf16(false, A, false, b3, (short)0, a3, false, false);
        }
        __syncthreads();    // all waves done reading hS; its space becomes gS

        // ---- spill gate accumulators to LDS (C layout: VGPR r, lane half) ----
#pragma unroll
        for (int r = 0; r < 8; ++r) {
            int row = mtile * 16 + khalf * 8 + r;
            int cb  = ngrp * 64;
            gS[row * NLOC + cb +  0 + nlane] = a0[r];
            gS[row * NLOC + cb + 16 + nlane] = a1[r];
            gS[row * NLOC + cb + 32 + nlane] = a2[r];
            gS[row * NLOC + cb + 48 + nlane] = a3[r];
        }
        __syncthreads();

        // ---- pointwise LSTM cell update; c lives in registers ----
        const int nxt = cur ^ 1;
#pragma unroll
        for (int j = 0; j < 8; ++j) {
            int pi = tid + j * NTHR;            // 0..2047 = 64 batch x 32 hcols
            int b  = pi >> 5;
            int hc = pi & 31;
            float fg = gS[b * NLOC +  0 + hc] + bsum[0 * H_ + wg * HCOLS + hc];
            float ig = gS[b * NLOC + 32 + hc] + bsum[1 * H_ + wg * HCOLS + hc];
            float mg = gS[b * NLOC + 64 + hc] + bsum[2 * H_ + wg * HCOLS + hc];
            float og = gS[b * NLOC + 96 + hc] + bsum[3 * H_ + wg * HCOLS + hc];
            float f = sigm(fg), i = sigm(ig), m = tanhf(mg), o = sigm(og);
            c[j] = f * c[j] + i * m;
            float hn = o * tanhf(c[j]);
            hbuf[(size_t)nxt * B_ * H_ + b * H_ + wg * HCOLS + hc] = f2bf(hn);
        }
        __threadfence();
        __syncthreads();

        // ---- device-scope barrier across the 16 persistent WGPs ----
        if (tid == 0) {
            __hip_atomic_fetch_add(bar, 1, __ATOMIC_ACQ_REL, __HIP_MEMORY_SCOPE_AGENT);
            const int tgt = NWG * (t + 1);
            while (__hip_atomic_load(bar, __ATOMIC_ACQUIRE, __HIP_MEMORY_SCOPE_AGENT) < tgt)
                __builtin_amdgcn_s_sleep(2);
        }
        __syncthreads();
        cur = nxt;
    }

    // ---- final FC: out[64,128] = h_T @ fc_w^T + fc_b (blocks 0..7, waves 0..3) ----
    if (wg < 8 && wave < 4) {
        const unsigned short* hT = hbuf + (size_t)cur * B_ * H_;
        const int n0 = wg * 16;
        v8f acc = {};
        const unsigned short* hrow = hT  + (size_t)arow * H_ + khalf * 16;
        const unsigned short* frow = fcw + (size_t)(n0 + nlane) * H_ + khalf * 16;
#pragma unroll 4
        for (int kb = 0; kb < H_ / 32; ++kb) {
            v16bf A  = *(const v16bf*)(hrow + kb * 32);
            v16bf Bm = *(const v16bf*)(frow + kb * 32);
            acc = __builtin_amdgcn_wmma_f32_16x16x32_bf16(false, A, false, Bm, (short)0, acc, false, false);
        }
#pragma unroll
        for (int r = 0; r < 8; ++r) {
            int row = mtile * 16 + khalf * 8 + r;
            int col = n0 + nlane;
            out[row * O_ + col] = acc[r] + fcb[col];
        }
    }
}

// ---------------- host launcher ----------------
extern "C" void kernel_launch(void* const* d_in, const int* in_sizes, int n_in,
                              void* d_out, int out_size, void* d_ws, size_t ws_size,
                              hipStream_t stream) {
    const float* x    = (const float*)d_in[0];
    const float* Wi   = (const float*)d_in[1];
    const float* bi   = (const float*)d_in[2];
    const float* Wh   = (const float*)d_in[3];
    const float* bh   = (const float*)d_in[4];
    const float* fcwf = (const float*)d_in[5];
    const float* fcb  = (const float*)d_in[6];
    float* out = (float*)d_out;

    char* ws = (char*)d_ws;
    unsigned short* Wbf   = (unsigned short*)(ws + OFF_W);
    unsigned short* fcwbf = (unsigned short*)(ws + OFF_FCW);
    float*          bsum  = (float*)(ws + OFF_BSUM);
    unsigned short* xTbf  = (unsigned short*)(ws + OFF_XT);
    unsigned short* hbuf  = (unsigned short*)(ws + OFF_HBUF);
    int*            bar   = (int*)(ws + OFF_BAR);

    k_prep_w<<<(G_ * K_) / 256, 256, 0, stream>>>(Wi, Wh, Wbf);
    k_prep_misc<<<(O_ * H_) / 256, 256, 0, stream>>>(bi, bh, fcwf, fcwbf, bsum, hbuf, bar);
    k_prep_x<<<(B_ * T_ * I_) / 256, 256, 0, stream>>>(x, xTbf);
    k_lstm<<<NWG, NTHR, (int)LDS_TOTAL, stream>>>(Wbf, bsum, xTbf, hbuf, bar, fcwbf, fcb, out);
}
